// DCNv2_57131654971627
// MI455X (gfx1250) — compile-verified, matrix operand order checked by reading, not run on hardware
//
#include <hip/hip_runtime.h>

typedef __attribute__((ext_vector_type(16))) _Float16 v16h;
typedef __attribute__((ext_vector_type(8)))  float    v8f;

#define B_    4
#define CIN_  64
#define H_    128
#define W_    128
#define COUT_ 64
#define KK_   9
#define HO_   128
#define WO_   128
#define R_    576      // CIN*9 reduction length
#define KT_   18       // 576 / 32
#define MT_   4        // COUT / 16
#define POSB_ 32       // output positions per block
#define NT_   2        // POSB / 16

static __device__ __forceinline__ int iclamp(int v, int lo, int hi) {
    return v < lo ? lo : (v > hi ? hi : v);
}

// ---------------------------------------------------------------------------
// Pre-pass: pack weight (f32, [COUT][CIN][3][3]) into A-fragment-layout f16:
//   aw[mt][kt][lane][i]  (16-bit A-matrix layout from CDNA5 ISA:
//   lane = M + 16*khalf; elem i<8 -> kk = khalf*8+i; i>=8 -> kk = 16+khalf*8+(i-8))
// ---------------------------------------------------------------------------
__global__ void dcn_pack_weight(const float* __restrict__ w, _Float16* __restrict__ aw) {
    int t = blockIdx.x * blockDim.x + threadIdx.x;
    if (t >= MT_ * KT_ * 32 * 16) return;
    int i    = t & 15;
    int lane = (t >> 4) & 31;
    int q    = t >> 9;          // mt*KT_ + kt
    int kt   = q % KT_;
    int mt   = q / KT_;
    int khalf = lane >> 4;
    int kk = (i < 8) ? (khalf * 8 + i) : (16 + khalf * 8 + (i - 8));
    int r  = kt * 32 + kk;                 // r = cin*9 + k == flat weight index
    int o  = mt * 16 + (lane & 15);
    aw[t] = (_Float16)w[(size_t)o * R_ + r];
}

// ---------------------------------------------------------------------------
// Fused deformable-sampling + WMMA GEMM.
// Block = 256 threads (8 wave32). One block = one (b, ho, 32-wide wo strip).
// ---------------------------------------------------------------------------
__global__ __launch_bounds__(256)
void dcn_main(const float* __restrict__ x, const float* __restrict__ off,
              const float* __restrict__ mask, const _Float16* __restrict__ aw,
              const float* __restrict__ bias, float* __restrict__ out) {
    // B-fragment-layout column buffer: [kt][ntile][lane][16 halves] = 36 KB
    __shared__ v16h colbuf[KT_ * NT_ * 32];

    const int tid  = threadIdx.x;
    const int lane = tid & 31;
    const int wave = tid >> 5;

    const int blk     = blockIdx.x;          // 0..2047
    const int wtile   = blk & 3;
    const int ho      = (blk >> 2) & (HO_ - 1);
    const int b       = blk >> 9;
    const int wo_base = wtile * POSB_;

    // ---------------- phase 1: deformable im2col ----------------
    {
        const int p  = lane;                 // position within strip
        const int wo = wo_base + p;
        const size_t sp = (size_t)ho * WO_ + wo;
        const float* xb = x + (size_t)b * CIN_ * H_ * W_;
        _Float16* cb = (_Float16*)colbuf;

        for (int k = wave; k < KK_; k += 8) {   // wave 0 handles k=0 and k=8
            const int kh = k / 3, kw = k % 3;
            const float oy = off[((size_t)b * (2 * KK_) + 2 * k    ) * (HO_ * WO_) + sp];
            const float ox = off[((size_t)b * (2 * KK_) + 2 * k + 1) * (HO_ * WO_) + sp];
            const float mk = mask[((size_t)b * KK_ + k) * (HO_ * WO_) + sp];

            const float py = (float)(ho - 1 + kh) + oy;
            const float px = (float)(wo - 1 + kw) + ox;
            const float y0f = floorf(py), x0f = floorf(px);
            const float ly = py - y0f, lx = px - x0f;
            const int y0 = (int)y0f, x0 = (int)x0f;
            const int y1 = y0 + 1,   x1 = x0 + 1;
            const float vy0 = (y0 >= 0 && y0 < H_) ? 1.f : 0.f;
            const float vy1 = (y1 >= 0 && y1 < H_) ? 1.f : 0.f;
            const float vx0 = (x0 >= 0 && x0 < W_) ? 1.f : 0.f;
            const float vx1 = (x1 >= 0 && x1 < W_) ? 1.f : 0.f;
            const int y0c = iclamp(y0, 0, H_ - 1), y1c = iclamp(y1, 0, H_ - 1);
            const int x0c = iclamp(x0, 0, W_ - 1), x1c = iclamp(x1, 0, W_ - 1);
            const int i00 = y0c * W_ + x0c, i01 = y0c * W_ + x1c;
            const int i10 = y1c * W_ + x0c, i11 = y1c * W_ + x1c;
            const float w00 = (1.f - ly) * (1.f - lx) * vy0 * vx0 * mk;
            const float w01 = (1.f - ly) * lx         * vy0 * vx1 * mk;
            const float w10 = ly         * (1.f - lx) * vy1 * vx0 * mk;
            const float w11 = ly         * lx         * vy1 * vx1 * mk;

            const int ntile = p >> 4;
            const int col   = p & 15;

            for (int c = 0; c < CIN_; ++c) {
                const float* xc = xb + (size_t)c * (H_ * W_);
                const float v = w00 * xc[i00] + w01 * xc[i01]
                              + w10 * xc[i10] + w11 * xc[i11];
                const int r     = c * KK_ + k;
                const int kt    = r >> 5;
                const int kk    = r & 31;
                const int khalf = (kk >> 3) & 1;
                const int elem  = (kk & 7) + ((kk & 16) >> 1);   // +8 if kk>=16
                const int flane = col + (khalf << 4);
                cb[(((kt * NT_ + ntile) * 32) + flane) * 16 + elem] = (_Float16)v;
            }
        }
    }
    __syncthreads();

    // ---------------- phase 2: WMMA GEMM ----------------
    {
        const int mtile = wave >> 1;
        const int ntile = wave & 1;
        v8f acc = {};
        const v16h* ap = (const v16h*)aw + (size_t)mtile * KT_ * 32 + lane;

        #pragma unroll
        for (int kt = 0; kt < KT_; ++kt) {
            v16h a = ap[(size_t)kt * 32];                         // coalesced 32B/lane
            v16h bfr = colbuf[(kt * NT_ + ntile) * 32 + lane];    // contiguous LDS
            acc = __builtin_amdgcn_wmma_f32_16x16x32_f16(
                      false, a, false, bfr, (short)0, acc, false, false);
        }

        const int col = lane & 15;
        const int mh  = lane >> 4;
        const int wo  = wo_base + ntile * 16 + col;
        #pragma unroll
        for (int v = 0; v < 8; ++v) {
            const int m  = v + 8 * mh;        // C/D layout: VGPR v -> row m (+8 for hi lanes)
            const int co = mtile * 16 + m;
            out[(((size_t)b * COUT_ + co) * HO_ + ho) * WO_ + wo] = acc[v] + bias[co];
        }
    }
}

// ---------------------------------------------------------------------------
extern "C" void kernel_launch(void* const* d_in, const int* in_sizes, int n_in,
                              void* d_out, int out_size, void* d_ws, size_t ws_size,
                              hipStream_t stream) {
    const float* x    = (const float*)d_in[0];
    const float* off  = (const float*)d_in[1];
    const float* mask = (const float*)d_in[2];
    const float* w    = (const float*)d_in[3];
    const float* bias = (const float*)d_in[4];
    float* out        = (float*)d_out;

    _Float16* aw = (_Float16*)d_ws;   // 4*18*32*16 halves = 73,728 bytes

    const int packN = MT_ * KT_ * 32 * 16;
    dcn_pack_weight<<<(packN + 255) / 256, 256, 0, stream>>>(w, aw);

    const int nblocks = B_ * HO_ * (WO_ / POSB_);   // 2048
    dcn_main<<<nblocks, 256, 0, stream>>>(x, off, mask, aw, bias, out);
}
